// TokenEncoder_68384469287337
// MI455X (gfx1250) — matrix operand order repaired and checked by reading, not compiled, not run on hardware
//
#include <hip/hip_runtime.h>
#include <hip/hip_bf16.h>

typedef unsigned short u16;
typedef unsigned int   u32;

typedef __attribute__((ext_vector_type(16))) __bf16 v16bf;
typedef __attribute__((ext_vector_type(8)))  float  v8f;

#define E_LAYERS 3
#define D_MODEL  512
#define D_INNER  512
#define D_STATE  64
#define D_FF     1024
#define DT_RANK  32
#define BB       4
#define LL       2048
#define NROW     (BB * LL)          // 8192 rows of activations

// ---------------- epilogue codes ----------------
#define EPI_NONE     0
#define EPI_RELU     1
#define EPI_SOFTPLUS 2

// ---------------- helpers ----------------
__device__ __forceinline__ u16 f2bf(float f) {
    u32 u = __float_as_uint(f);
    u32 r = u + 0x7FFFu + ((u >> 16) & 1u);
    return (u16)(r >> 16);
}
__device__ __forceinline__ float silu(float v) {
    return v / (1.0f + __expf(-v));
}

// f32 (possibly strided) -> compact bf16
__global__ void cvt_bf16_kernel(const float* __restrict__ src, int ld, int cols,
                                u16* __restrict__ dst, long long total) {
    long long i = (long long)blockIdx.x * blockDim.x + threadIdx.x;
    if (i >= total) return;
    long long r = i / cols, c = i % cols;
    dst[i] = f2bf(src[r * (long long)ld + c]);
}

// ---------------- WMMA GEMM:  C[M,N] = A[M,K] * W[N,K]^T (+bias)(+act)(+resid) ----------------
// Register-blocked: each wave computes a 16(M) x (NSUB*16)(N) strip.
// NSUB is a compile-time constant: fully unrolled, branch-free K-loop
// (no divergent guards around WMMA; EXEC stays all-ones; no spills).
union ABfrag { v16bf v; uint4 q[2]; };

template <int NSUB>
__global__ void gemm_wmma_bf16(const u16* __restrict__ A, int lda,
                               const u16* __restrict__ W, int ldw,
                               const float* __restrict__ bias,
                               const float* __restrict__ resid,
                               float* __restrict__ C, int ldc,
                               int colBase, int tilesN, int totalTiles,
                               int K, int epi) {
    const int wave = threadIdx.x >> 5;
    const int lane = threadIdx.x & 31;
    const int tileId = blockIdx.x * (blockDim.x >> 5) + wave;
    if (tileId >= totalTiles) return;                  // wave-uniform exit

    const int tm = tileId / tilesN;
    const int tn = tileId - tm * tilesN;
    const int m0 = tm << 4;
    const int n0 = colBase + tn * (NSUB << 4);
    const int half = lane >> 4;                        // 0: lanes 0-15, 1: lanes 16-31
    const int l15  = lane & 15;

    // A fragment: row m, K chunks per ISA layout (halves own K+0..7/16..23 vs +8..15/24..31)
    const u16* arow  = A + (size_t)(m0 + l15) * lda + (half << 3);
    // B fragment (= W row n): same K chunks as A (halves own K+0..7/16..23 vs +8..15/24..31)
    const u16* brow0 = W + (size_t)(n0 + l15) * ldw + (half << 3);
    const size_t bstep = (size_t)16 * ldw;             // next 16-col subtile of W

    v8f acc[NSUB];
#pragma unroll
    for (int j = 0; j < NSUB; ++j) acc[j] = (v8f){};

    for (int k0 = 0; k0 < K; k0 += 32) {
        ABfrag a;
        a.q[0] = *(const uint4*)(arow + k0);           // K = kb+0..kb+7
        a.q[1] = *(const uint4*)(arow + k0 + 16);      // K = kb+16..kb+23
#pragma unroll
        for (int j = 0; j < NSUB; ++j) {
            ABfrag b;
            const u16* bp = brow0 + (size_t)j * bstep + k0;
            b.q[0] = *(const uint4*)(bp);              // K = kb+0..kb+7
            b.q[1] = *(const uint4*)(bp + 16);         // K = kb+16..kb+23
            acc[j] = __builtin_amdgcn_wmma_f32_16x16x32_bf16(
                false, a.v, false, b.v, (short)0, acc[j], false, false);
        }
    }

    // D layout: VGPR r -> row m0 + r + half*8, col (subtile base) + l15
    const int cmBase = m0 + (half << 3);
#pragma unroll
    for (int j = 0; j < NSUB; ++j) {
        const int cn = n0 + (j << 4) + l15;
        const float bv = bias ? bias[cn] : 0.0f;
#pragma unroll
        for (int r = 0; r < 8; ++r) {
            float v = acc[j][r] + bv;
            if (epi == EPI_RELU) {
                v = fmaxf(v, 0.0f);
            } else if (epi == EPI_SOFTPLUS) {
                v = (v > 20.0f) ? v : log1pf(__expf(v));
            }
            size_t idx = (size_t)(cmBase + r) * ldc + cn;
            if (resid) v += resid[idx];
            C[idx] = v;
        }
    }
}

// ---------------- depthwise causal conv (k=2) + bias + SiLU ----------------
__global__ void conv_silu_kernel(const float* __restrict__ xz,
                                 const float* __restrict__ cw,
                                 const float* __restrict__ cb,
                                 float* __restrict__ u) {
    long long i = (long long)blockIdx.x * blockDim.x + threadIdx.x;
    if (i >= (long long)NROW * D_INNER) return;
    int d = (int)(i % D_INNER);
    long long row = i / D_INNER;           // b*L + t
    int t = (int)(row % LL);
    float cur  = xz[row * (2 * D_INNER) + d];
    float prev = (t > 0) ? xz[(row - 1) * (2 * D_INNER) + d] : 0.0f;
    float v = cw[d * 2 + 1] * cur + cw[d * 2 + 0] * prev + cb[d];
    u[i] = silu(v);
}

// ---------------- selective scan (sequential over L), fused silu(z) gate ----------------
// grid: (D_INNER/8, B), block 256 (8 waves). Wave w owns channel d = bx*8+w.
// Lane k owns states n=k and n=k+32. Loads for step t+1 are issued before the
// exp/shfl chain of step t (software pipelining hides L2 latency on the
// serial dependence chain).
__global__ void scan_kernel(const float* __restrict__ u,
                            const float* __restrict__ delta,
                            const float* __restrict__ dbl,
                            const float* __restrict__ A_log,
                            const float* __restrict__ Dv,
                            const float* __restrict__ xz,
                            float* __restrict__ y) {
    const int wave = threadIdx.x >> 5;
    const int lane = threadIdx.x & 31;
    const int d = blockIdx.x * 8 + wave;
    const int b = blockIdx.y;

    const float A0 = -__expf(A_log[d * D_STATE + lane]);
    const float A1 = -__expf(A_log[d * D_STATE + lane + 32]);
    const float Dd = Dv[d];
    float h0 = 0.0f, h1 = 0.0f;

    const size_t row0 = (size_t)b * LL;
    // prologue: load t = 0
    float dt_c = delta[row0 * D_INNER + d];
    float ut_c = u[row0 * D_INNER + d];
    const float* bc0 = dbl + row0 * (DT_RANK + 2 * D_STATE);
    float B0c = bc0[DT_RANK + lane];
    float B1c = bc0[DT_RANK + lane + 32];
    float C0c = bc0[DT_RANK + D_STATE + lane];
    float C1c = bc0[DT_RANK + D_STATE + lane + 32];
    float z_c = xz[row0 * (2 * D_INNER) + D_INNER + d];

    for (int t = 0; t < LL; ++t) {
        // issue next-step loads first (overlap with this step's compute)
        float dt_n = 0.0f, ut_n = 0.0f, B0n = 0.0f, B1n = 0.0f,
              C0n = 0.0f, C1n = 0.0f, z_n = 0.0f;
        if (t + 1 < LL) {
            size_t rn = row0 + t + 1;
            dt_n = delta[rn * D_INNER + d];
            ut_n = u[rn * D_INNER + d];
            const float* bcn = dbl + rn * (DT_RANK + 2 * D_STATE);
            B0n = bcn[DT_RANK + lane];
            B1n = bcn[DT_RANK + lane + 32];
            C0n = bcn[DT_RANK + D_STATE + lane];
            C1n = bcn[DT_RANK + D_STATE + lane + 32];
            z_n = xz[rn * (2 * D_INNER) + D_INNER + d];
        }

        float du = dt_c * ut_c;
        h0 = h0 * __expf(dt_c * A0) + du * B0c;
        h1 = h1 * __expf(dt_c * A1) + du * B1c;
        float acc = h0 * C0c + h1 * C1c;
#pragma unroll
        for (int off = 16; off > 0; off >>= 1)
            acc += __shfl_xor(acc, off, 32);
        if (lane == 0) {
            float yy = acc + ut_c * Dd;
            y[(row0 + t) * D_INNER + d] = yy * silu(z_c);
        }

        dt_c = dt_n; ut_c = ut_n;
        B0c = B0n; B1c = B1n; C0c = C0n; C1c = C1n; z_c = z_n;
    }
}

// ---------------- layer norm over d=512 (one block per row) ----------------
__global__ void layernorm_kernel(const float* __restrict__ s,
                                 const float* __restrict__ g,
                                 const float* __restrict__ bta,
                                 float* __restrict__ out) {
    const int row = blockIdx.x;
    const int tid = threadIdx.x;               // 256 threads, 512 cols
    const float* sr = s + (size_t)row * D_MODEL;
    float x0 = sr[tid], x1 = sr[tid + 256];
    float sum = x0 + x1;
    float sq  = x0 * x0 + x1 * x1;
#pragma unroll
    for (int off = 16; off > 0; off >>= 1) {
        sum += __shfl_xor(sum, off, 32);
        sq  += __shfl_xor(sq,  off, 32);
    }
    __shared__ float ssum[8], ssq[8];
    int wv = tid >> 5, ln = tid & 31;
    if (ln == 0) { ssum[wv] = sum; ssq[wv] = sq; }
    __syncthreads();
    if (tid == 0) {
        float ts = 0.0f, tq = 0.0f;
        for (int i = 0; i < 8; ++i) { ts += ssum[i]; tq += ssq[i]; }
        ssum[0] = ts; ssq[0] = tq;
    }
    __syncthreads();
    float mean = ssum[0] * (1.0f / D_MODEL);
    float var  = ssq[0] * (1.0f / D_MODEL) - mean * mean;
    float inv  = rsqrtf(var + 1e-5f);
    float* o = out + (size_t)row * D_MODEL;
    o[tid]       = (x0 - mean) * inv * g[tid]       + bta[tid];
    o[tid + 256] = (x1 - mean) * inv * g[tid + 256] + bta[tid + 256];
}

// ---------------- host-side launch helpers ----------------
static inline void cvt_launch(const float* src, int ld, int cols, u16* dst,
                              long long rows, hipStream_t s) {
    long long total = rows * cols;
    int blocks = (int)((total + 255) / 256);
    hipLaunchKernelGGL(cvt_bf16_kernel, dim3(blocks), dim3(256), 0, s, src, ld, cols, dst, total);
}

// Splits N into a 64-wide main region (NSUB=4) and a 16/32/48-wide remainder.
static inline void gemm_launch(const u16* A, int lda, const u16* W, int ldw,
                               const float* bias, const float* resid,
                               float* C, int ldc, int M, int N, int K, int epi,
                               hipStream_t s) {
    const int tilesM = M / 16;
    const int nFull = (N / 64) * 64;
    if (nFull > 0) {
        int tilesN = nFull / 64;
        int tiles = tilesM * tilesN;
        int blocks = (tiles + 3) / 4;          // 4 waves (128 threads) per block
        hipLaunchKernelGGL((gemm_wmma_bf16<4>), dim3(blocks), dim3(128), 0, s,
                           A, lda, W, ldw, bias, resid, C, ldc,
                           0, tilesN, tiles, K, epi);
    }
    const int rem = N - nFull;                 // multiple of 16
    if (rem > 0) {
        int tiles = tilesM;                    // tilesN == 1
        int blocks = (tiles + 3) / 4;
        switch (rem / 16) {
        case 1:
            hipLaunchKernelGGL((gemm_wmma_bf16<1>), dim3(blocks), dim3(128), 0, s,
                               A, lda, W, ldw, bias, resid, C, ldc,
                               nFull, 1, tiles, K, epi);
            break;
        case 2:
            hipLaunchKernelGGL((gemm_wmma_bf16<2>), dim3(blocks), dim3(128), 0, s,
                               A, lda, W, ldw, bias, resid, C, ldc,
                               nFull, 1, tiles, K, epi);
            break;
        default:
            hipLaunchKernelGGL((gemm_wmma_bf16<3>), dim3(blocks), dim3(128), 0, s,
                               A, lda, W, ldw, bias, resid, C, ldc,
                               nFull, 1, tiles, K, epi);
            break;
        }
    }
}

// ---------------- workspace layout (float units) ----------------
#define OFF_XCUR 0LL
#define OFF_XZ   (OFF_XCUR + (long long)NROW * D_MODEL)        //  4,194,304
#define OFF_U    (OFF_XZ   + (long long)NROW * 2 * D_INNER)    // 12,582,912
#define OFF_DBL  (OFF_U    + (long long)NROW * D_INNER)        // 16,777,216
#define OFF_DEL  (OFF_DBL  + (long long)NROW * (DT_RANK + 2 * D_STATE))
#define OFF_Y    (OFF_DEL  + (long long)NROW * D_INNER)
#define OFF_SUM  (OFF_Y    + (long long)NROW * D_INNER)
#define OFF_ABF  (OFF_SUM  + (long long)NROW * D_MODEL)        // bf16 staging (8192*1024 u16)
#define OFF_WBF  (OFF_ABF  + (long long)NROW * 2 * D_INNER / 2)

// bf16 weight offsets (u16 units), full (E,...) tensors
#define WPL_IN  (2 * D_MODEL * D_MODEL)                    // 524288 per layer
#define WPL_XP  ((DT_RANK + 2 * D_STATE) * D_INNER)        //  81920
#define WPL_DT  (D_INNER * DT_RANK)                        //  16384
#define WPL_OUT (D_MODEL * D_INNER)                        // 262144
#define WPL_F1  (D_FF * D_MODEL)                           // 524288
#define WPL_F2  (D_MODEL * D_FF)                           // 524288
#define WB_IN   0LL
#define WB_XP   (WB_IN  + (long long)E_LAYERS * WPL_IN)
#define WB_DT   (WB_XP  + (long long)E_LAYERS * WPL_XP)
#define WB_OUT  (WB_DT  + (long long)E_LAYERS * WPL_DT)
#define WB_F1   (WB_OUT + (long long)E_LAYERS * WPL_OUT)
#define WB_F2   (WB_F1  + (long long)E_LAYERS * WPL_F1)

extern "C" void kernel_launch(void* const* d_in, const int* in_sizes, int n_in,
                              void* d_out, int out_size, void* d_ws, size_t ws_size,
                              hipStream_t stream) {
    const float* x_in   = (const float*)d_in[0];
    const float* in_w   = (const float*)d_in[1];
    const float* conv_w = (const float*)d_in[2];
    const float* conv_b = (const float*)d_in[3];
    const float* xp_w   = (const float*)d_in[4];
    const float* dtp_w  = (const float*)d_in[5];
    const float* dtp_b  = (const float*)d_in[6];
    const float* A_log  = (const float*)d_in[7];
    const float* Dv     = (const float*)d_in[8];
    const float* out_w  = (const float*)d_in[9];
    const float* ln1_g  = (const float*)d_in[10];
    const float* ln1_b  = (const float*)d_in[11];
    const float* ffn_w1 = (const float*)d_in[12];
    const float* ffn_b1 = (const float*)d_in[13];
    const float* ffn_w2 = (const float*)d_in[14];
    const float* ffn_b2 = (const float*)d_in[15];
    const float* ln2_g  = (const float*)d_in[16];
    const float* ln2_b  = (const float*)d_in[17];
    float* out = (float*)d_out;

    float* ws    = (float*)d_ws;
    float* xcur  = ws + OFF_XCUR;
    float* xz    = ws + OFF_XZ;
    float* u     = ws + OFF_U;
    float* dbl   = ws + OFF_DBL;
    float* delta = ws + OFF_DEL;
    float* yb    = ws + OFF_Y;
    float* sum   = ws + OFF_SUM;
    u16*   abf   = (u16*)(ws + OFF_ABF);
    u16*   wbf   = (u16*)(ws + OFF_WBF);

    // Per-launch bf16 weight conversion — deterministic, re-done every call.
    cvt_launch(in_w,   D_MODEL, D_MODEL, wbf + WB_IN,  (long long)E_LAYERS * 2 * D_MODEL, stream);
    cvt_launch(xp_w,   D_INNER, D_INNER, wbf + WB_XP,  (long long)E_LAYERS * (DT_RANK + 2 * D_STATE), stream);
    cvt_launch(dtp_w,  DT_RANK, DT_RANK, wbf + WB_DT,  (long long)E_LAYERS * D_INNER, stream);
    cvt_launch(out_w,  D_INNER, D_INNER, wbf + WB_OUT, (long long)E_LAYERS * D_MODEL, stream);
    cvt_launch(ffn_w1, D_MODEL, D_MODEL, wbf + WB_F1,  (long long)E_LAYERS * D_FF, stream);
    cvt_launch(ffn_w2, D_FF,    D_FF,    wbf + WB_F2,  (long long)E_LAYERS * D_MODEL, stream);

    hipMemcpyAsync(xcur, x_in, (size_t)NROW * D_MODEL * sizeof(float),
                   hipMemcpyDeviceToDevice, stream);

    for (int l = 0; l < E_LAYERS; ++l) {
        // ---- mamba block ----
        // xz = x @ in_w^T            [8192,1024]
        cvt_launch(xcur, D_MODEL, D_MODEL, abf, NROW, stream);
        gemm_launch(abf, D_MODEL, wbf + WB_IN + (long long)l * WPL_IN, D_MODEL,
                    nullptr, nullptr, xz, 2 * D_INNER, NROW, 2 * D_INNER, D_MODEL,
                    EPI_NONE, stream);
        // u = silu(causal_conv(xc) + cb)
        {
            long long total = (long long)NROW * D_INNER;
            int blocks = (int)((total + 255) / 256);
            hipLaunchKernelGGL(conv_silu_kernel, dim3(blocks), dim3(256), 0, stream,
                               xz, conv_w + (long long)l * D_INNER * 2,
                               conv_b + (long long)l * D_INNER, u);
        }
        // dbl = u @ xp_w^T           [8192,160]
        cvt_launch(u, D_INNER, D_INNER, abf, NROW, stream);
        gemm_launch(abf, D_INNER, wbf + WB_XP + (long long)l * WPL_XP, D_INNER,
                    nullptr, nullptr, dbl, DT_RANK + 2 * D_STATE,
                    NROW, DT_RANK + 2 * D_STATE, D_INNER, EPI_NONE, stream);
        // delta = softplus(dt @ dtp_w^T + dtp_b)
        cvt_launch(dbl, DT_RANK + 2 * D_STATE, DT_RANK, abf, NROW, stream);
        gemm_launch(abf, DT_RANK, wbf + WB_DT + (long long)l * WPL_DT, DT_RANK,
                    dtp_b + (long long)l * D_INNER, nullptr, delta, D_INNER,
                    NROW, D_INNER, DT_RANK, EPI_SOFTPLUS, stream);
        // selective scan + silu(z) gate -> yb
        hipLaunchKernelGGL(scan_kernel, dim3(D_INNER / 8, BB), dim3(256), 0, stream,
                           u, delta, dbl,
                           A_log + (long long)l * D_INNER * D_STATE,
                           Dv + (long long)l * D_INNER, yb == nullptr ? nullptr : xz, yb);
        // sum = x + y @ out_w^T ; x = LN1(sum)
        cvt_launch(yb, D_INNER, D_INNER, abf, NROW, stream);
        gemm_launch(abf, D_INNER, wbf + WB_OUT + (long long)l * WPL_OUT, D_INNER,
                    nullptr, xcur, sum, D_MODEL, NROW, D_MODEL, D_INNER,
                    EPI_NONE, stream);
        hipLaunchKernelGGL(layernorm_kernel, dim3(NROW), dim3(256), 0, stream,
                           sum, ln1_g + (long long)l * D_MODEL,
                           ln1_b + (long long)l * D_MODEL, xcur);

        // ---- FFN ----  (xz buffer reused as hidden h)
        cvt_launch(xcur, D_MODEL, D_MODEL, abf, NROW, stream);
        gemm_launch(abf, D_MODEL, wbf + WB_F1 + (long long)l * WPL_F1, D_MODEL,
                    ffn_b1 + (long long)l * D_FF, nullptr, xz, D_FF,
                    NROW, D_FF, D_MODEL, EPI_RELU, stream);
        cvt_launch(xz, D_FF, D_FF, abf, NROW, stream);
        gemm_launch(abf, D_FF, wbf + WB_F2 + (long long)l * WPL_F2, D_FF,
                    ffn_b2 + (long long)l * D_MODEL, xcur, sum, D_MODEL,
                    NROW, D_MODEL, D_FF, EPI_NONE, stream);
        float* dest = (l == E_LAYERS - 1) ? out : xcur;
        hipLaunchKernelGGL(layernorm_kernel, dim3(NROW), dim3(256), 0, stream,
                           sum, ln2_g + (long long)l * D_MODEL,
                           ln2_b + (long long)l * D_MODEL, dest);
    }
}
